// MultiHeadSelfAttention_11931419149053
// MI455X (gfx1250) — matrix operand order repaired and checked by reading, not compiled, not run on hardware
//
#include <hip/hip_runtime.h>

#define S_LEN 4096
#define D_MODEL 1024
#define NHEAD 16
#define DK 64

#if __has_builtin(__builtin_amdgcn_global_load_async_to_lds_b128) && \
    __has_builtin(__builtin_amdgcn_s_wait_asynccnt)
#define USE_ASYNC_LDS 1
#else
#define USE_ASYNC_LDS 0
#endif

typedef __attribute__((ext_vector_type(16))) __bf16 v16bf;
typedef __attribute__((ext_vector_type(8)))  float  v8f;
typedef __attribute__((ext_vector_type(4)))  int    v4i;

union BF16Frag {
    v16bf bf;
    float4 f4[2];
    unsigned short us[16];
};

__device__ __forceinline__ unsigned short f2bf(float f) {
    unsigned int u = __builtin_bit_cast(unsigned int, f);
    unsigned int r = u + 0x7FFFu + ((u >> 16) & 1u);   // round-to-nearest-even
    return (unsigned short)(r >> 16);
}

// A fragment: 16 rows (m) x 32 K from row-major [rows][ld] bf16.
// lane<16 -> row=m0+lane, K {0..7,16..23}; lane>=16 -> row=m0+lane-16, K {8..15,24..31}
__device__ __forceinline__ BF16Frag load_frag_a(const unsigned short* base, int ld,
                                                int m0, int kbase, int lane) {
    int lo = lane & 15, hi = lane >> 4;
    const unsigned short* p = base + (size_t)(m0 + lo) * ld + kbase + hi * 8;
    BF16Frag f;
    f.f4[0] = *(const float4*)(p);
    f.f4[1] = *(const float4*)(p + 16);
    return f;
}

// B fragment: 32 K x 16 N from "N-major" [ncols][ld] bf16 (B transposed).
// lane<16 -> N=n0+lane, K {0..7,16..23}; lane>=16 -> N=n0+lane-16, K {8..15,24..31}
__device__ __forceinline__ BF16Frag load_frag_b(const unsigned short* base, int ld,
                                                int n0, int kbase, int lane) {
    int lo = lane & 15, hi = lane >> 4;
    const unsigned short* p = base + (size_t)(n0 + lo) * ld + kbase + hi * 8;
    BF16Frag f;
    f.f4[0] = *(const float4*)(p);
    f.f4[1] = *(const float4*)(p + 16);
    return f;
}

__device__ __forceinline__ v8f wmma_bf16(const BF16Frag& a, const BF16Frag& b, v8f c) {
    return __builtin_amdgcn_wmma_f32_16x16x32_bf16(false, a.bf, false, b.bf,
                                                   (short)0, c, false, false);
}

// Cooperative 16B-per-thread global->LDS copy (async path if toolchain exposes it).
__device__ __forceinline__ void copy16_to_lds(const unsigned short* g, unsigned short* l) {
#if USE_ASYNC_LDS
    __builtin_amdgcn_global_load_async_to_lds_b128(
        (__attribute__((address_space(1))) v4i*)g,
        (__attribute__((address_space(3))) v4i*)l, 0, 0);
#else
    *(float4*)l = *(const float4*)g;
#endif
}

__device__ __forceinline__ void copy_fence() {
#if USE_ASYNC_LDS
    __builtin_amdgcn_s_wait_asynccnt(0);
#endif
}

// ---------------- conversion kernels ----------------

__global__ void __launch_bounds__(256) cvt_f32_bf16(const float* __restrict__ in,
                                                    unsigned short* __restrict__ out, int n) {
    int i = blockIdx.x * 256 + threadIdx.x;
    if (i < n) out[i] = f2bf(in[i]);
}

// in: [rows][cols] f32 ; out: [cols][rows] bf16  (Wt[n][k] = W[k][n])
__global__ void __launch_bounds__(256) transpose_f32_bf16(const float* __restrict__ in,
                                                          unsigned short* __restrict__ out,
                                                          int rows, int cols) {
    int i = blockIdx.x * 256 + threadIdx.x;
    if (i < rows * cols) {
        int k = i / cols, n = i - k * cols;
        out[(size_t)n * rows + k] = f2bf(in[i]);
    }
}

// ---------------- WMMA GEMM: C[M,N] = A[M,K] * Bt[N,K]^T + bias ----------------
// 64x32 register tile per wave (4 A-frags x 2 B-frags, 8 WMMA per K=32 chunk).
// out_mode 0: fp32 row-major [M][N]
// out_mode 1: bf16 head-major [h][row][dk] with scale   (Q / K)
// out_mode 2: bf16 head-transposed [h][dk][row]         (V)
__global__ void __launch_bounds__(256) gemm_bf16_kernel(
    const unsigned short* __restrict__ A, const unsigned short* __restrict__ Bt,
    const float* __restrict__ bias, float scale, int M, int N, int K, int out_mode,
    float* __restrict__ outf, unsigned short* __restrict__ outb, int rows_total)
{
    int wave = threadIdx.x >> 5;
    int lane = threadIdx.x & 31;
    int tile = blockIdx.x * 8 + wave;
    int tiles_n = N >> 5;
    int tm = tile / tiles_n, tn = tile - tm * tiles_n;
    int m0 = tm << 6, n0 = tn << 5;

    v8f acc[4][2] = {};
    for (int kc = 0; kc < K; kc += 32) {
        BF16Frag a[4];
        #pragma unroll
        for (int mi = 0; mi < 4; mi++)
            a[mi] = load_frag_a(A, K, m0 + mi * 16, kc, lane);
        BF16Frag b0 = load_frag_b(Bt, K, n0, kc, lane);
        BF16Frag b1 = load_frag_b(Bt, K, n0 + 16, kc, lane);
        #pragma unroll
        for (int mi = 0; mi < 4; mi++) {
            acc[mi][0] = wmma_bf16(a[mi], b0, acc[mi][0]);
            acc[mi][1] = wmma_bf16(a[mi], b1, acc[mi][1]);
        }
    }

    int lo = lane & 15, hi = lane >> 4;
    #pragma unroll
    for (int mi = 0; mi < 4; mi++) {
        #pragma unroll
        for (int ni = 0; ni < 2; ni++) {
            #pragma unroll
            for (int r = 0; r < 8; r++) {
                int row = m0 + mi * 16 + r + (hi << 3);
                int col = n0 + ni * 16 + lo;
                float v = acc[mi][ni][r] + bias[col];
                if (out_mode == 0) {
                    outf[(size_t)row * N + col] = v;
                } else if (out_mode == 1) {
                    outb[((size_t)(col >> 6) * rows_total + row) * DK + (col & 63)] = f2bf(v * scale);
                } else {
                    outb[(size_t)(col >> 6) * DK * rows_total + (size_t)(col & 63) * rows_total + row] = f2bf(v);
                }
            }
        }
    }
}

// ---------------- flash attention (causal) ----------------
// Block = 8 waves = one head x 256 consecutive queries (32 queries per wave).
// K/V key blocks (32 keys) staged cooperatively in LDS once per iteration and
// consumed by all 8 waves (8x cut in global K/V traffic vs per-wave loads).
// Q,K: bf16 [H][S][DK] (Q pre-scaled by 1/sqrt(DK));  Vt: bf16 [H][DK][S]
// O: bf16 [S][D] merged-head row-major
__global__ void __launch_bounds__(256) attn_kernel(
    const unsigned short* __restrict__ Q, const unsigned short* __restrict__ K,
    const unsigned short* __restrict__ Vt, unsigned short* __restrict__ O)
{
    __shared__ __align__(16) unsigned short kbuf[32 * DK];      // 4 KB: K rows kb..kb+31
    __shared__ __align__(16) unsigned short vbuf[DK * 32];      // 4 KB: Vt rows 0..63, keys kb..kb+31
    __shared__ __align__(16) unsigned short plds[8][32 * 32];   // 16 KB: per-wave P tiles

    int tid = threadIdx.x;
    int wave = tid >> 5, lane = tid & 31;
    int lo = lane & 15, hi = lane >> 4;

    int h = blockIdx.x >> 4;              // 16 query-blocks per head
    int qbase = (blockIdx.x & 15) << 8;   // 256 queries per block
    int m0 = qbase + (wave << 5);         // this wave: queries m0..m0+31

    const unsigned short* Qh = Q  + (size_t)h * S_LEN * DK;
    const unsigned short* Kh = K  + (size_t)h * S_LEN * DK;
    const unsigned short* Vh = Vt + (size_t)h * DK * S_LEN;
    unsigned short* pw = plds[wave];

    BF16Frag qa[2][2];
    #pragma unroll
    for (int rt = 0; rt < 2; rt++)
        #pragma unroll
        for (int c = 0; c < 2; c++)
            qa[rt][c] = load_frag_a(Qh, DK, m0 + rt * 16, c * 32, lane);

    v8f o[2][4] = {};
    float mrun[2][8], lrun[2][8];
    #pragma unroll
    for (int rt = 0; rt < 2; rt++)
        #pragma unroll
        for (int r = 0; r < 8; r++) { mrun[rt][r] = -1e30f; lrun[rt][r] = 0.0f; }

    // cooperative staging indices: 256 threads x 16B cover each 4KB panel
    int vrow = tid >> 2, vcol = (tid & 3) << 3;

    const int kb_end = qbase + 256;   // block-uniform causal bound
    for (int kb = 0; kb < kb_end; kb += 32) {
        // stage K block (contiguous 4KB) and V block (64 rows x 64B)
        copy16_to_lds(Kh + (size_t)kb * DK + (tid << 3), kbuf + (tid << 3));
        copy16_to_lds(Vh + (size_t)vrow * S_LEN + kb + vcol, vbuf + vrow * 32 + vcol);
        copy_fence();
        __syncthreads();

        if (kb <= m0 + 31) {   // wave-uniform causal skip
            // scores: S[32x32] = Q[32x64] * K_blk^T
            BF16Frag kf[2][2];
            #pragma unroll
            for (int t = 0; t < 2; t++)
                #pragma unroll
                for (int c = 0; c < 2; c++)
                    kf[t][c] = load_frag_b(kbuf, DK, t * 16, c * 32, lane);

            v8f s[2][2] = {};
            #pragma unroll
            for (int rt = 0; rt < 2; rt++)
                #pragma unroll
                for (int t = 0; t < 2; t++) {
                    s[rt][t] = wmma_bf16(qa[rt][0], kf[t][0], s[rt][t]);
                    s[rt][t] = wmma_bf16(qa[rt][1], kf[t][1], s[rt][t]);
                }

            const bool need_mask = (kb + 31 > m0);
            #pragma unroll
            for (int rt = 0; rt < 2; rt++) {
                if (need_mask) {
                    #pragma unroll
                    for (int r = 0; r < 8; r++) {
                        int qrow = m0 + rt * 16 + r + (hi << 3);
                        if (kb + lo > qrow)      s[rt][0][r] = -1e30f;
                        if (kb + 16 + lo > qrow) s[rt][1][r] = -1e30f;
                    }
                }
                // online softmax: each row lives across 16 lanes of a half-wave
                #pragma unroll
                for (int r = 0; r < 8; r++) {
                    float rm = fmaxf(s[rt][0][r], s[rt][1][r]);
                    rm = fmaxf(rm, __shfl_xor(rm, 1, 32));
                    rm = fmaxf(rm, __shfl_xor(rm, 2, 32));
                    rm = fmaxf(rm, __shfl_xor(rm, 4, 32));
                    rm = fmaxf(rm, __shfl_xor(rm, 8, 32));
                    float mnew  = fmaxf(mrun[rt][r], rm);
                    float alpha = __expf(mrun[rt][r] - mnew);
                    float p0 = __expf(s[rt][0][r] - mnew);
                    float p1 = __expf(s[rt][1][r] - mnew);
                    float rs = p0 + p1;
                    rs += __shfl_xor(rs, 1, 32);
                    rs += __shfl_xor(rs, 2, 32);
                    rs += __shfl_xor(rs, 4, 32);
                    rs += __shfl_xor(rs, 8, 32);
                    lrun[rt][r] = lrun[rt][r] * alpha + rs;
                    mrun[rt][r] = mnew;
                    #pragma unroll
                    for (int j = 0; j < 4; j++) o[rt][j][r] *= alpha;
                    int prow = rt * 16 + r + (hi << 3);
                    pw[prow * 32 + lo]      = f2bf(p0);
                    pw[prow * 32 + 16 + lo] = f2bf(p1);
                }
            }

            // reload P as A-fragments (D-layout -> A-layout via LDS)
            BF16Frag pf[2];
            #pragma unroll
            for (int rt = 0; rt < 2; rt++) {
                const unsigned short* pp = pw + (rt * 16 + lo) * 32 + hi * 8;
                pf[rt].f4[0] = *(const float4*)(pp);
                pf[rt].f4[1] = *(const float4*)(pp + 16);
            }

            // O[32x64] += P[32x32] * V_blk[32x64]
            BF16Frag vf[4];
            #pragma unroll
            for (int j = 0; j < 4; j++)
                vf[j] = load_frag_b(vbuf, 32, j * 16, 0, lane);
            #pragma unroll
            for (int rt = 0; rt < 2; rt++)
                #pragma unroll
                for (int j = 0; j < 4; j++)
                    o[rt][j] = wmma_bf16(pf[rt], vf[j], o[rt][j]);
        }
        __syncthreads();
    }

    // normalize and store merged-head O (bf16 [S][D])
    #pragma unroll
    for (int rt = 0; rt < 2; rt++) {
        #pragma unroll
        for (int r = 0; r < 8; r++) {
            float inv = 1.0f / lrun[rt][r];
            int row = m0 + rt * 16 + r + (hi << 3);
            size_t base = (size_t)row * D_MODEL + h * DK + lo;
            O[base + 0]  = f2bf(o[rt][0][r] * inv);
            O[base + 16] = f2bf(o[rt][1][r] * inv);
            O[base + 32] = f2bf(o[rt][2][r] * inv);
            O[base + 48] = f2bf(o[rt][3][r] * inv);
        }
    }
}

// ---------------- host launch ----------------

extern "C" void kernel_launch(void* const* d_in, const int* in_sizes, int n_in,
                              void* d_out, int out_size, void* d_ws, size_t ws_size,
                              hipStream_t stream) {
    (void)in_sizes; (void)n_in; (void)out_size; (void)ws_size;
    const float* x  = (const float*)d_in[0];
    // d_in[1] = causal mask (implemented analytically)
    const float* Wq = (const float*)d_in[2];
    const float* bq = (const float*)d_in[3];
    const float* Wk = (const float*)d_in[4];
    const float* bk = (const float*)d_in[5];
    const float* Wv = (const float*)d_in[6];
    const float* bv = (const float*)d_in[7];
    const float* Wo = (const float*)d_in[8];
    const float* bo = (const float*)d_in[9];
    float* out = (float*)d_out;

    char* ws = (char*)d_ws;
    const size_t MB = 1024ull * 1024ull;
    unsigned short* xb  = (unsigned short*)(ws + 0);        //  8 MB  x bf16 [S][D]
    unsigned short* wqt = (unsigned short*)(ws + 8  * MB);  //  2 MB  Wq^T bf16
    unsigned short* wkt = (unsigned short*)(ws + 10 * MB);
    unsigned short* wvt = (unsigned short*)(ws + 12 * MB);
    unsigned short* wot = (unsigned short*)(ws + 14 * MB);
    unsigned short* Qb  = (unsigned short*)(ws + 16 * MB);  //  8 MB  [H][S][DK]
    unsigned short* Kb  = (unsigned short*)(ws + 24 * MB);  //  8 MB  [H][S][DK]
    unsigned short* Vtb = (unsigned short*)(ws + 32 * MB);  //  8 MB  [H][DK][S]
    unsigned short* Ob  = (unsigned short*)(ws + 40 * MB);  //  8 MB  [S][D]

    const int SD = S_LEN * D_MODEL;   // 4M
    const int DD = D_MODEL * D_MODEL; // 1M

    cvt_f32_bf16<<<SD / 256, 256, 0, stream>>>(x, xb, SD);
    transpose_f32_bf16<<<DD / 256, 256, 0, stream>>>(Wq, wqt, D_MODEL, D_MODEL);
    transpose_f32_bf16<<<DD / 256, 256, 0, stream>>>(Wk, wkt, D_MODEL, D_MODEL);
    transpose_f32_bf16<<<DD / 256, 256, 0, stream>>>(Wv, wvt, D_MODEL, D_MODEL);
    transpose_f32_bf16<<<DD / 256, 256, 0, stream>>>(Wo, wot, D_MODEL, D_MODEL);

    // (S/64)*(D/32) = 64*32 = 2048 wave-tiles -> 256 blocks of 8 waves
    const int gemm_blocks = (S_LEN / 64) * (D_MODEL / 32) / 8;
    const float qscale = 0.125f;  // 1/sqrt(DK)
    gemm_bf16_kernel<<<gemm_blocks, 256, 0, stream>>>(xb, wqt, bq, qscale, S_LEN, D_MODEL, D_MODEL, 1, nullptr, Qb,  S_LEN);
    gemm_bf16_kernel<<<gemm_blocks, 256, 0, stream>>>(xb, wkt, bk, 1.0f,   S_LEN, D_MODEL, D_MODEL, 1, nullptr, Kb,  S_LEN);
    gemm_bf16_kernel<<<gemm_blocks, 256, 0, stream>>>(xb, wvt, bv, 1.0f,   S_LEN, D_MODEL, D_MODEL, 2, nullptr, Vtb, S_LEN);

    // 16 heads * 16 query-blocks = 256 blocks of 8 waves (32 queries/wave)
    attn_kernel<<<NHEAD * (S_LEN / 256), 256, 0, stream>>>(Qb, Kb, Vtb, Ob);

    gemm_bf16_kernel<<<gemm_blocks, 256, 0, stream>>>(Ob, wot, bo, 1.0f, S_LEN, D_MODEL, D_MODEL, 0, out, nullptr, S_LEN);
}
